// AttentionGRUCell_75960791597430
// MI455X (gfx1250) — matrix operand order repaired and checked by reading, not compiled, not run on hardware
//
#include <hip/hip_runtime.h>

// ---------------------------------------------------------------------------
// AttentionGRUCell for MI455X (gfx1250): bf16 WMMA GEMM pipeline with
// double-buffered GLOBAL_LOAD_ASYNC_TO_LDS staging (ASYNCcnt-tracked).
// B=512, T=64, XD=512, ED=1024, U=1024
// ---------------------------------------------------------------------------

typedef __attribute__((ext_vector_type(16))) __bf16     v16bf;
typedef __attribute__((ext_vector_type(8)))  float      v8f;
typedef __attribute__((ext_vector_type(4)))  unsigned   u32x4;

union FragBF {
    u32x4 u4[2];
    v16bf v;
};

__device__ __forceinline__ unsigned short f2bf(float f) {
    unsigned u = __float_as_uint(f);
    u += 0x7FFFu + ((u >> 16) & 1u);     // round-to-nearest-even
    return (unsigned short)(u >> 16);
}

__device__ __forceinline__ float hsig(float x) {
    return fminf(fmaxf(0.2f * x + 0.5f, 0.0f), 1.0f);
}

// gfx1250 async global->LDS copy, 16B per lane. dst: wave-relative LDS byte
// offset; src: global byte address.
__device__ __forceinline__ void async_copy_b128(unsigned lds_off, unsigned long long gaddr) {
    asm volatile("global_load_async_to_lds_b128 %0, %1, off"
                 :: "v"(lds_off), "v"(gaddr)
                 : "memory");
}
#define ASYNC_WAIT(n) asm volatile("s_wait_asynccnt " #n ::: "memory")

__device__ __forceinline__ unsigned lds_off32(const void* p) {
    return (unsigned)(size_t)p;          // low 32 bits = LDS offset
}

// ------------------------------ prep kernels ------------------------------

__global__ void k_tanh_cvt(const float* __restrict__ src,
                           unsigned short* __restrict__ dst, int n) {
    int i = blockIdx.x * 256 + threadIdx.x;
    if (i < n) dst[i] = f2bf(tanhf(src[i]));
}

__global__ void k_cvt(const float* __restrict__ src,
                      unsigned short* __restrict__ dst, int n) {
    int i = blockIdx.x * 256 + threadIdx.x;
    if (i < n) dst[i] = f2bf(src[i]);
}

// src: R x C (f32, row-major)  ->  dst: C x R (bf16, row-major)  (i.e. src^T)
__global__ void k_transpose_cvt(const float* __restrict__ src,
                                unsigned short* __restrict__ dst,
                                int R, int C, int n) {
    int i = blockIdx.x * 256 + threadIdx.x;
    if (i >= n) return;
    int r = i / C, c = i % C;
    dst[(size_t)c * R + r] = f2bf(src[i]);
}

__global__ void k_zero(float* __restrict__ p, int n) {
    int i = blockIdx.x * 256 + threadIdx.x;
    if (i < n) p[i] = 0.0f;
}

// ------------------------------ WMMA GEMM ---------------------------------
// C[M,N] = A[M,K](bf16) * Bt[N,K](bf16)^T, f32 accumulate.
// Block tile 128(M) x 128(N), 8 waves (wave32) in 4x2; each wave 32x64
// (2x4 WMMA tiles). K step 32 (v_wmma_f32_16x16x32_bf16).
// Tiles staged with async global->LDS copies, double buffered; per-thread
// global addresses are strided pointers bumped by 64B per K tile.

#define BM 128
#define BN 128
#define BK 32
#define LDT 40                   // halves per LDS tile row (80B, 16B-aligned)
#define BUFB (BM * LDT * 2)      // LDS bytes per (A or B) buffer = 10240

struct Stager {
    unsigned long long gA0, gA1, gB0, gB1;   // global byte addresses (tile 0)
    unsigned lA0, lA1, lB0, lB1;             // LDS byte offsets (buffer 0)

    __device__ __forceinline__ void init(
        const unsigned short* A, int lda, int bM,
        const unsigned short* Bt, int ldb, int bN,
        const unsigned short* sA0, const unsigned short* sB0, int tid)
    {
        int r0 = tid >> 2,          cc0 = tid & 3;
        int r1 = (tid + 256) >> 2,  cc1 = tid & 3;
        gA0 = (unsigned long long)(size_t)(A + (size_t)(bM + r0) * lda + cc0 * 8);
        gA1 = (unsigned long long)(size_t)(A + (size_t)(bM + r1) * lda + cc1 * 8);
        gB0 = (unsigned long long)(size_t)(Bt + (size_t)(bN + r0) * ldb + cc0 * 8);
        gB1 = (unsigned long long)(size_t)(Bt + (size_t)(bN + r1) * ldb + cc1 * 8);
        lA0 = lds_off32(sA0 + r0 * LDT + cc0 * 8);
        lA1 = lds_off32(sA0 + r1 * LDT + cc1 * 8);
        lB0 = lds_off32(sB0 + r0 * LDT + cc0 * 8);
        lB1 = lds_off32(sB0 + r1 * LDT + cc1 * 8);
    }

    // issue 4 async ops (per wave) into buffer `buf`, advance to next K tile
    __device__ __forceinline__ void stage(int buf) {
        unsigned lo = buf ? (unsigned)BUFB : 0u;
        async_copy_b128(lA0 + lo, gA0);
        async_copy_b128(lA1 + lo, gA1);
        async_copy_b128(lB0 + lo, gB0);
        async_copy_b128(lB1 + lo, gB1);
        gA0 += BK * 2; gA1 += BK * 2; gB0 += BK * 2; gB1 += BK * 2;
    }
};

__device__ __forceinline__ void load_frags_and_mma(
    const unsigned short* sA, const unsigned short* sB,
    int waveM, int waveN, int half, int l16, v8f acc[2][4])
{
    FragBF a[2], b[4];
    // A frag (ISA 7.12.2, 16-bit A 16x32): halves 0-7 = K[half*8..+8),
    // halves 8-15 = K[16+half*8..+8) of row (l16)
#pragma unroll
    for (int mi = 0; mi < 2; ++mi) {
        const unsigned short* p = sA + (waveM + mi * 16 + l16) * LDT + half * 8;
        a[mi].u4[0] = *(const u32x4*)p;
        a[mi].u4[1] = *(const u32x4*)(p + 16);
    }
    // B frag: halves 0-7 = K[half*8..+8), halves 8-15 = K[16+half*8..+8) of column (l16)
#pragma unroll
    for (int ni = 0; ni < 4; ++ni) {
        const unsigned short* p = sB + (waveN + ni * 16 + l16) * LDT + half * 8;
        b[ni].u4[0] = *(const u32x4*)p;
        b[ni].u4[1] = *(const u32x4*)(p + 16);
    }
#pragma unroll
    for (int mi = 0; mi < 2; ++mi)
#pragma unroll
        for (int ni = 0; ni < 4; ++ni)
            acc[mi][ni] = __builtin_amdgcn_wmma_f32_16x16x32_bf16(
                false, a[mi].v, false, b[ni].v,
                (short)0, acc[mi][ni], false, false);
}

// Core double-buffered K loop shared by both GEMM kernels.
#define GEMM_MAIN_LOOP()                                                      \
    v8f acc[2][4] = {};                                                       \
    const int nt = K / BK;                                                    \
    Stager st;                                                                \
    st.init(A, lda, bM, Bt, ldb, bN, sA[0], sB[0], tid);                      \
    st.stage(0);                                                              \
    for (int kt = 0; kt < nt - 1; ++kt) {                                     \
        int cur = kt & 1;                                                     \
        st.stage(cur ^ 1);                                                    \
        ASYNC_WAIT(4);          /* tile kt landed (4 newest = tile kt+1) */   \
        __syncthreads();                                                      \
        load_frags_and_mma(sA[cur], sB[cur], waveM, waveN, half, l16, acc);   \
        __syncthreads();        /* all reads done before buffer reuse */      \
    }                                                                         \
    ASYNC_WAIT(0);                                                            \
    __syncthreads();                                                          \
    load_frags_and_mma(sA[(nt - 1) & 1], sB[(nt - 1) & 1],                    \
                       waveM, waveN, half, l16, acc);

__global__ __launch_bounds__(256) void k_gemm(
    const unsigned short* __restrict__ A, int lda,
    const unsigned short* __restrict__ Bt, int ldb,
    float* __restrict__ C, int ldc, int K)
{
    __shared__ __align__(16) unsigned short sA[2][BM * LDT];
    __shared__ __align__(16) unsigned short sB[2][BN * LDT];

    const int tid  = threadIdx.x;
    const int lane = tid & 31;
    const int wave = tid >> 5;
    const int half = lane >> 4;
    const int l16  = lane & 15;
    const int bM = blockIdx.y * BM;
    const int bN = blockIdx.x * BN;
    const int waveM = (wave & 3) * 32;
    const int waveN = (wave >> 2) * 64;

    GEMM_MAIN_LOOP();

    // C layout: vgpr j -> row = 8*half + j, col = l16
#pragma unroll
    for (int mi = 0; mi < 2; ++mi)
#pragma unroll
        for (int ni = 0; ni < 4; ++ni) {
            int row0 = bM + waveM + mi * 16 + half * 8;
            int col  = bN + waveN + ni * 16 + l16;
#pragma unroll
            for (int j = 0; j < 8; ++j)
                C[(size_t)(row0 + j) * ldc + col] = acc[mi][ni][j];
        }
}

// Attention GEMM: acc = tanh_enc @ WaT_enc^T; epilogue fuses
// e[row] += sum_n tanh(A_h[b][n] + acc[row][n]) * Va[n]   (row = b*T + t)
__global__ __launch_bounds__(256) void k_gemm_att(
    const unsigned short* __restrict__ A, int lda,
    const unsigned short* __restrict__ Bt, int ldb,
    const float* __restrict__ Ah,   // (512 x 1024) f32
    const float* __restrict__ Va,   // (1024) f32
    float* __restrict__ e,          // (32768) f32, pre-zeroed
    int K)
{
    __shared__ __align__(16) unsigned short sA[2][BM * LDT];
    __shared__ __align__(16) unsigned short sB[2][BN * LDT];

    const int tid  = threadIdx.x;
    const int lane = tid & 31;
    const int wave = tid >> 5;
    const int half = lane >> 4;
    const int l16  = lane & 15;
    const int bM = blockIdx.y * BM;
    const int bN = blockIdx.x * BN;
    const int waveM = (wave & 3) * 32;
    const int waveN = (wave >> 2) * 64;

    GEMM_MAIN_LOOP();

    // fused epilogue: per-row tanh(Ah + acc)*Va partial dot, half-wave reduce
#pragma unroll
    for (int mi = 0; mi < 2; ++mi) {
#pragma unroll
        for (int j = 0; j < 8; ++j) {
            int row  = bM + waveM + mi * 16 + half * 8 + j;  // = b*64 + t
            int bidx = row >> 6;
            float s = 0.0f;
#pragma unroll
            for (int ni = 0; ni < 4; ++ni) {
                int col = bN + waveN + ni * 16 + l16;
                float v = acc[mi][ni][j] + Ah[(size_t)bidx * 1024 + col];
                s += tanhf(v) * Va[col];
            }
#pragma unroll
            for (int m = 1; m < 16; m <<= 1)
                s += __shfl_xor(s, m, 32);
            if (l16 == 0) atomicAdd(&e[row], s);
        }
    }
}

// ------------------------- attention tail kernels -------------------------

__global__ void k_softmax64(const float* __restrict__ e,
                            float* __restrict__ alpha,
                            float* __restrict__ out) {
    int b = blockIdx.x, t = threadIdx.x;
    __shared__ float s[64];
    float v = e[b * 64 + t];
    s[t] = v; __syncthreads();
    for (int o = 32; o > 0; o >>= 1) { if (t < o) s[t] = fmaxf(s[t], s[t + o]); __syncthreads(); }
    float mx = s[0]; __syncthreads();
    float ex = __expf(v - mx);
    s[t] = ex; __syncthreads();
    for (int o = 32; o > 0; o >>= 1) { if (t < o) s[t] += s[t + o]; __syncthreads(); }
    float a = ex / s[0];
    alpha[b * 64 + t]  = a;
    out[b * 1088 + t]  = a;
}

__global__ void k_ctx(const float* __restrict__ alpha,
                      const float* __restrict__ enc,
                      float* __restrict__ ct) {
    int b = blockIdx.x;
    __shared__ float sa[64];
    if (threadIdx.x < 64) sa[threadIdx.x] = alpha[b * 64 + threadIdx.x];
    __syncthreads();
    for (int c = threadIdx.x; c < 1024; c += 256) {
        const float* p = enc + (size_t)b * 64 * 1024 + c;
        float s = 0.0f;
#pragma unroll 8
        for (int t = 0; t < 64; ++t) s += sa[t] * p[t * 1024];
        ct[b * 1024 + c] = s;
    }
}

__global__ void k_buildx(const float* __restrict__ inputs,
                         const float* __restrict__ ct,
                         unsigned short* __restrict__ xb) {
    int i = blockIdx.x * 256 + threadIdx.x;          // 512*1536
    if (i >= 512 * 1536) return;
    int b = i / 1536, c = i % 1536;
    float v = (c < 512) ? inputs[b * 512 + c] : ct[b * 1024 + (c - 512)];
    xb[i] = f2bf(v);
}

__global__ void k_gates(const float* __restrict__ Gx,
                        const float* __restrict__ Gh,
                        const float* __restrict__ bias,
                        const float* __restrict__ h,
                        float* __restrict__ out) {
    int i = blockIdx.x * 256 + threadIdx.x;          // 512*1024
    if (i >= 512 * 1024) return;
    int b = i >> 10, n = i & 1023;
    const float* gx = Gx + (size_t)b * 3072;
    const float* gh = Gh + (size_t)b * 3072;
    float z  = hsig(gx[n]        + gh[n]        + bias[n]);
    float r  = hsig(gx[n + 1024] + gh[n + 1024] + bias[n + 1024]);
    float hh = tanhf(gx[n + 2048] + bias[n + 2048] + r * gh[n + 2048]);
    float hv = h[i];
    out[b * 1088 + 64 + n] = z * hv + (1.0f - z) * hh;
}

// ------------------------------ launcher ----------------------------------

extern "C" void kernel_launch(void* const* d_in, const int* in_sizes, int n_in,
                              void* d_out, int out_size, void* d_ws, size_t ws_size,
                              hipStream_t stream) {
    const float* inputs = (const float*)d_in[0];   // (512, 512)
    const float* h      = (const float*)d_in[1];   // (512, 1024)
    const float* enc    = (const float*)d_in[2];   // (512, 64, 1024)
    const float* Wk     = (const float*)d_in[3];   // (1536, 3072)
    const float* Wr     = (const float*)d_in[4];   // (1024, 3072)
    const float* bias   = (const float*)d_in[5];   // (3072,)
    const float* Wa     = (const float*)d_in[6];   // (2048, 1024)
    const float* Va     = (const float*)d_in[7];   // (1024, 1)
    float* out = (float*)d_out;                    // (512, 1088)

    char* w = (char*)d_ws;
    unsigned short* enc_t = (unsigned short*)(w);              // 32768x1024 bf16
    unsigned short* th    = (unsigned short*)(w + 67108864);   // 512x1024 bf16 tanh(h)
    unsigned short* hb    = (unsigned short*)(w + 68157440);   // 512x1024 bf16 h
    unsigned short* WaT   = (unsigned short*)(w + 69206016);   // 1024x2048 bf16 (Wa^T)
    unsigned short* kT    = (unsigned short*)(w + 73400320);   // 3072x1536 bf16 (kernel^T)
    unsigned short* rT    = (unsigned short*)(w + 82837504);   // 3072x1024 bf16 (recurrent^T)
    float* Ah    = (float*)(w + 89128960);                     // 512x1024
    float* e     = (float*)(w + 91226112);                     // 512x64
    float* alpha = (float*)(w + 91357184);                     // 512x64
    float* ct    = (float*)(w + 91488256);                     // 512x1024
    unsigned short* xb = (unsigned short*)(w + 93585408);      // 512x1536 bf16
    float* Gx    = (float*)(w + 95158272);                     // 512x3072
    float* Gh    = (float*)(w + 101449728);                    // 512x3072

    // ---- prep: weight transpose+cvt, activation tanh+cvt ----
    k_transpose_cvt<<<dim3((2048 * 1024 + 255) / 256), 256, 0, stream>>>(Wa, WaT, 2048, 1024, 2048 * 1024);
    k_transpose_cvt<<<dim3((1536 * 3072 + 255) / 256), 256, 0, stream>>>(Wk, kT, 1536, 3072, 1536 * 3072);
    k_transpose_cvt<<<dim3((1024 * 3072 + 255) / 256), 256, 0, stream>>>(Wr, rT, 1024, 3072, 1024 * 3072);
    k_tanh_cvt<<<dim3(33554432 / 256), 256, 0, stream>>>(enc, enc_t, 33554432);
    k_tanh_cvt<<<dim3(524288 / 256), 256, 0, stream>>>(h, th, 524288);
    k_cvt<<<dim3(524288 / 256), 256, 0, stream>>>(h, hb, 524288);
    k_zero<<<dim3(32768 / 256), 256, 0, stream>>>(e, 32768);

    // ---- attention scores ----
    // A_h = tanh(h) @ Wa[:1024,:]   (M=512, N=1024, K=1024)
    k_gemm<<<dim3(1024 / BN, 512 / BM), 256, 0, stream>>>(th, 1024, WaT, 2048, Ah, 1024, 1024);
    // e += sum_n tanh(A_h + tanh(enc) @ Wa[1024:,:]) * Va  (M=32768, N=1024, K=1024)
    k_gemm_att<<<dim3(1024 / BN, 32768 / BM), 256, 0, stream>>>(enc_t, 1024, WaT + 1024, 2048, Ah, Va, e, 1024);

    // ---- softmax + context ----
    k_softmax64<<<dim3(512), 64, 0, stream>>>(e, alpha, out);
    k_ctx<<<dim3(512), 256, 0, stream>>>(alpha, enc, ct);

    // ---- GRU gates ----
    k_buildx<<<dim3(512 * 1536 / 256), 256, 0, stream>>>(inputs, ct, xb);
    // Gx = x @ kernel   (M=512, N=3072, K=1536)
    k_gemm<<<dim3(3072 / BN, 512 / BM), 256, 0, stream>>>(xb, 1536, kT, 1536, Gx, 3072, 1536);
    // Gh = h @ recurrent (M=512, N=3072, K=1024)
    k_gemm<<<dim3(3072 / BN, 512 / BM), 256, 0, stream>>>(hb, 1024, rT, 1024, Gh, 3072, 1024);
    k_gates<<<dim3(512 * 1024 / 256), 256, 0, stream>>>(Gx, Gh, bias, h, out);
}
